// LinearAttention_57002805952957
// MI455X (gfx1250) — compile-verified, matrix operand order checked by reading, not run on hardware
//
#include <hip/hip_runtime.h>
#include <hip/hip_bf16.h>
#include <math.h>

// ---------------------------------------------------------------------------
// Problem constants (match reference)
// ---------------------------------------------------------------------------
#define Bq 2
#define Lq 2048
#define Dq 2048
#define HKq 16
#define HVq 32
#define DKq 128
#define DVq 128
#define KEY_DIM 2048      // HK*DK
#define VAL_DIM 4096      // HV*DV
#define CONV_DIM 8192     // 2*KEY_DIM + VAL_DIM
#define NTOK 4096         // B*L
#define EPSq 1e-6f

typedef __attribute__((ext_vector_type(16))) __bf16 v16bf;
typedef __attribute__((ext_vector_type(8)))  float  v8f;

union FragBF {
    v16bf v;
    unsigned int u[8];
};

__device__ __forceinline__ unsigned short f2bf(float f) {
    unsigned int u = __float_as_uint(f);
    unsigned int r = u + 0x7FFFu + ((u >> 16) & 1u);   // round-to-nearest-even
    return (unsigned short)(r >> 16);
}
__device__ __forceinline__ unsigned int pack_bf2(float lo, float hi) {
    return (unsigned int)f2bf(lo) | ((unsigned int)f2bf(hi) << 16);
}

// ---------------------------------------------------------------------------
// WMMA bf16 GEMM:  C[M,N] = A[M,K] * B[K,N]   (fp32 in/out, bf16 compute)
// block = 256 threads (8 waves), tile 128x128, BK = 64 (2 WMMA K-steps)
// LDS holds bf16 pairs packed in dwords:
//   AsU: [m][32]  dword q = (K pair 2q,2q+1) of row m
//   BsU: [n][32]  dword q = (K pair 2q,2q+1) of column n   (K-transposed)
// ---------------------------------------------------------------------------
#define BM 128
#define BN 128
#define BK 64

__device__ __forceinline__ void load_frag_u(const unsigned int* __restrict__ smem,
                                            int row, int kkHalf, FragBF& f) {
    // ISA 16-bit A/B fragment layout (16x32 bf16):
    //   lanes 0-15 : row = lane,     K-base 0   (dword base 0)
    //   lanes 16-31: row = lane-16,  K-base 8   (dword base 4)
    //   VGPR j<4 : K = kbase+2j     -> dword kbase/2 + j
    //   VGPR j>=4: K = 16+kbase+2j' -> dword 8 + kbase/2 + j'
    int lane = threadIdx.x & 31;
    int r    = row + (lane & 15);
    int kb   = (lane >> 4) * 4;                  // kbase/2 in dwords
    const unsigned int* p = smem + r * 32 + kkHalf + kb;
#pragma unroll
    for (int j = 0; j < 4; ++j) f.u[j] = p[j];           // -> ds_load_b128
#pragma unroll
    for (int j = 0; j < 4; ++j) f.u[4 + j] = p[8 + j];   // -> ds_load_b128
}

__global__ __launch_bounds__(256) void gemm_bf16_wmma(const float* __restrict__ Ag,
                                                      const float* __restrict__ Bg,
                                                      float* __restrict__ Cg,
                                                      int M, int N, int K) {
    __shared__ __align__(16) unsigned int AsU[BM * (BK / 2)]; // 16 KB
    __shared__ __align__(16) unsigned int BsU[BN * (BK / 2)]; // 16 KB

    const int bm   = blockIdx.y;
    const int bn   = blockIdx.x;
    const int tid  = threadIdx.x;
    const int lane = tid & 31;
    const int wave = tid >> 5;      // 0..7
    const int wm = wave >> 2;       // 0..1  -> rows [wm*64, +64)
    const int wn = wave & 3;        // 0..3  -> cols [wn*32, +32)

    v8f acc[4][2] = {};

    for (int k0 = 0; k0 < K; k0 += BK) {
        // ---- stage A tile: 128x64 fp32 -> packed bf16 (b128 loads, b64 stores)
#pragma unroll
        for (int i = 0; i < 8; ++i) {
            int idx4 = tid + i * 256;            // 0..2047 float4's
            int m  = idx4 >> 4;                  // 16 float4 per row
            int k4 = idx4 & 15;
            float4 v = *reinterpret_cast<const float4*>(
                &Ag[(size_t)(bm * BM + m) * K + k0 + k4 * 4]);
            uint2 pk;
            pk.x = pack_bf2(v.x, v.y);
            pk.y = pack_bf2(v.z, v.w);
            *reinterpret_cast<uint2*>(&AsU[m * 32 + k4 * 2]) = pk;
        }
        // ---- stage B tile: 64x128 fp32 -> packed bf16, K-pair transposed
        //      item = (kp, n4): rows 2kp,2kp+1 at cols n4*4..+3
#pragma unroll
        for (int i = 0; i < 4; ++i) {
            int idx4 = tid + i * 256;            // 0..1023
            int kp = idx4 >> 5;                  // 0..31
            int n4 = idx4 & 31;
            const float* r0p = &Bg[(size_t)(k0 + 2 * kp) * N + bn * BN + n4 * 4];
            float4 r0 = *reinterpret_cast<const float4*>(r0p);
            float4 r1 = *reinterpret_cast<const float4*>(r0p + N);
            BsU[(n4 * 4 + 0) * 32 + kp] = pack_bf2(r0.x, r1.x);
            BsU[(n4 * 4 + 1) * 32 + kp] = pack_bf2(r0.y, r1.y);
            BsU[(n4 * 4 + 2) * 32 + kp] = pack_bf2(r0.z, r1.z);
            BsU[(n4 * 4 + 3) * 32 + kp] = pack_bf2(r0.w, r1.w);
        }
        // L2 prefetch of next K-tile (global_prefetch_b8 on gfx1250)
        if (k0 + BK < K) {
            int m = tid >> 1, k = (tid & 1) * 32;
            __builtin_prefetch(&Ag[(size_t)(bm * BM + m) * K + k0 + BK + k], 0, 1);
            __builtin_prefetch(&Bg[(size_t)(k0 + BK + (tid >> 2)) * N + bn * BN + (tid & 3) * 32], 0, 1);
        }
        __syncthreads();

#pragma unroll
        for (int kk = 0; kk < 2; ++kk) {         // two 16x16x32 K-steps
            FragBF af[4], bfr[2];
#pragma unroll
            for (int mt = 0; mt < 4; ++mt)
                load_frag_u(AsU, wm * 64 + mt * 16, kk * 16, af[mt]);
#pragma unroll
            for (int nt = 0; nt < 2; ++nt)
                load_frag_u(BsU, wn * 32 + nt * 16, kk * 16, bfr[nt]);
#pragma unroll
            for (int mt = 0; mt < 4; ++mt)
#pragma unroll
                for (int nt = 0; nt < 2; ++nt)
                    acc[mt][nt] = __builtin_amdgcn_wmma_f32_16x16x32_bf16(
                        false, af[mt].v, false, bfr[nt].v,
                        (short)0, acc[mt][nt], false, false);
        }
        __syncthreads();
    }

    // ---- epilogue: D layout -> lane = N col, VGPR j: M = j (+8 for lanes>=16)
    const int nloc = lane & 15;
    const int mhi  = (lane >> 4) * 8;
#pragma unroll
    for (int mt = 0; mt < 4; ++mt)
#pragma unroll
        for (int nt = 0; nt < 2; ++nt) {
            int n = bn * BN + wn * 32 + nt * 16 + nloc;
#pragma unroll
            for (int j = 0; j < 8; ++j) {
                int m = bm * BM + wm * 64 + mt * 16 + mhi + j;
                Cg[(size_t)m * N + n] = acc[mt][nt][j];
            }
        }
}

// ---------------------------------------------------------------------------
// beta = sigmoid(x @ W_b);  gate = -exp(A_log) * softplus(x @ W_a + dt_bias)
// 8 tokens x 32 heads per 256-thread block
// ---------------------------------------------------------------------------
__global__ __launch_bounds__(256) void proj_gate(const float* __restrict__ x,
                                                 const float* __restrict__ Wb,
                                                 const float* __restrict__ Wa,
                                                 const float* __restrict__ A_log,
                                                 const float* __restrict__ dt_bias,
                                                 float* __restrict__ beta,
                                                 float* __restrict__ gate) {
    int tid = threadIdx.x;
    int tok = blockIdx.x * 8 + (tid >> 5);
    int h   = tid & 31;
    const float* xr = x + (size_t)tok * Dq;
    float ab = 0.f, aa = 0.f;
    for (int k = 0; k < Dq; ++k) {
        float xv = xr[k];
        ab = fmaf(xv, Wb[k * HVq + h], ab);
        aa = fmaf(xv, Wa[k * HVq + h], aa);
    }
    beta[(size_t)tok * HVq + h] = 1.f / (1.f + expf(-ab));
    float s  = aa + dt_bias[h];
    float sp = (s > 20.f) ? s : log1pf(expf(s));
    gate[(size_t)tok * HVq + h] = -expf(A_log[h]) * sp;
}

// ---------------------------------------------------------------------------
// causal depthwise conv(K=4) + SiLU + split q/k/v + per-head L2 norm
// grid = (64 groups, L, B), block = 128 (one channel per thread)
// ---------------------------------------------------------------------------
__global__ __launch_bounds__(128) void conv_silu_split(const float* __restrict__ mixed,
                                                       const float* __restrict__ cw,
                                                       const float* __restrict__ cb,
                                                       float* __restrict__ qn,
                                                       float* __restrict__ kn,
                                                       float* __restrict__ vv) {
    const int g = blockIdx.x;      // channel group 0..63 (x128)
    const int l = blockIdx.y;
    const int b = blockIdx.z;
    const int d = threadIdx.x;     // 0..127
    const int c = g * 128 + d;

    float acc = cb[c];
#pragma unroll
    for (int kk = 0; kk < 4; ++kk) {
        int li = l + kk - 3;
        if (li >= 0)
            acc = fmaf(cw[c * 4 + kk],
                       mixed[((size_t)(b * Lq + li)) * CONV_DIM + c], acc);
    }
    float y = acc / (1.f + expf(-acc));   // SiLU

    __shared__ float red[128];
    if (g < 32) {                         // q or k head -> L2 normalize over 128
        red[d] = y * y;
        __syncthreads();
        for (int s = 64; s > 0; s >>= 1) {
            if (d < s) red[d] += red[d + s];
            __syncthreads();
        }
        float rn = rsqrtf(red[0] + EPSq);
        size_t tokbase = (size_t)(b * Lq + l);
        if (g < 16) {
            qn[(tokbase * HKq + g) * DKq + d] = y * rn * 0.08838834764831845f; // *DK^-0.5
        } else {
            kn[(tokbase * HKq + (g - 16)) * DKq + d] = y * rn;
        }
    } else {
        vv[((size_t)(b * Lq + l) * HVq + (g - 32)) * DVq + d] = y;
    }
}

// ---------------------------------------------------------------------------
// gated delta rule scan: one 256-thread block per (b, h_v); state in registers.
// thread layout: vcol = tid&127, half = tid>>7 owns k in [half*64, half*64+64)
// ---------------------------------------------------------------------------
__global__ __launch_bounds__(256) void gdr_scan(const float* __restrict__ qn,
                                                const float* __restrict__ kn,
                                                const float* __restrict__ vv,
                                                const float* __restrict__ gate,
                                                const float* __restrict__ beta,
                                                float* __restrict__ attn) {
    const int b  = blockIdx.x / HVq;
    const int h  = blockIdx.x % HVq;
    const int hk = h >> 1;                // GQA repeat factor 2
    const int tid  = threadIdx.x;
    const int vcol = tid & 127;
    const int k0   = (tid >> 7) * 64;

    __shared__ float q_s[128], k_s[128], v_s[128], delta_s[128];
    __shared__ float red[256];
    __shared__ float scal[2];

    float S[64];
#pragma unroll
    for (int j = 0; j < 64; ++j) S[j] = 0.f;

    for (int t = 0; t < Lq; ++t) {
        size_t tok   = (size_t)(b * Lq + t);
        size_t qbase = (tok * HKq + hk) * DKq;
        size_t vbase = (tok * HVq + h) * DVq;
        size_t gi    = tok * HVq + h;

        if (tid < 128) {
            q_s[tid] = qn[qbase + tid];
            v_s[tid] = vv[vbase + tid];
        } else {
            k_s[tid - 128] = kn[qbase + (tid - 128)];
        }
        if (tid == 0) { scal[0] = gate[gi]; scal[1] = beta[gi]; }
        __syncthreads();                                   // (1)

        float dec = expf(scal[0]);
        float acc = 0.f;
#pragma unroll
        for (int j = 0; j < 64; ++j) {
            S[j] *= dec;
            acc = fmaf(S[j], k_s[k0 + j], acc);
        }
        red[tid] = acc;
        __syncthreads();                                   // (2)

        if (tid < 128) {
            float vp = red[tid] + red[tid + 128];
            delta_s[tid] = scal[1] * (v_s[tid] - vp);
        }
        __syncthreads();                                   // (3)

        float dl = delta_s[vcol];
        float acc2 = 0.f;
#pragma unroll
        for (int j = 0; j < 64; ++j) {
            S[j] = fmaf(k_s[k0 + j], dl, S[j]);
            acc2 = fmaf(S[j], q_s[k0 + j], acc2);
        }
        red[tid] = acc2;
        __syncthreads();                                   // (4)

        if (tid < 128)
            attn[vbase + tid] = red[tid] + red[tid + 128];
        // staging arrays for t+1 are only rewritten after the barrier below, in the next
        // iteration; reads above all completed before barrier (4). safe.
        __syncthreads();
    }
}

// ---------------------------------------------------------------------------
// per-(token,head) RMS norm * norm_w, gated by SiLU(z). one row (128) / block.
// ---------------------------------------------------------------------------
__global__ __launch_bounds__(128) void norm_gate(const float* __restrict__ attn,
                                                 const float* __restrict__ z,
                                                 const float* __restrict__ nw,
                                                 float* __restrict__ hg) {
    const int r = blockIdx.x;
    const int d = threadIdx.x;
    float h = attn[(size_t)r * 128 + d];
    __shared__ float red[128];
    red[d] = h * h;
    __syncthreads();
    for (int s = 64; s > 0; s >>= 1) {
        if (d < s) red[d] += red[d + s];
        __syncthreads();
    }
    float var = red[0] * (1.f / 128.f);
    float y = h * rsqrtf(var + EPSq) * nw[d];
    float zv = z[(size_t)r * 128 + d];
    hg[(size_t)r * 128 + d] = y * (zv / (1.f + expf(-zv)));
}

// ---------------------------------------------------------------------------
// Launch
// ---------------------------------------------------------------------------
extern "C" void kernel_launch(void* const* d_in, const int* in_sizes, int n_in,
                              void* d_out, int out_size, void* d_ws, size_t ws_size,
                              hipStream_t stream) {
    const float* x       = (const float*)d_in[0];
    const float* W_qkv   = (const float*)d_in[1];
    const float* W_z     = (const float*)d_in[2];
    const float* W_b     = (const float*)d_in[3];
    const float* W_a     = (const float*)d_in[4];
    const float* A_log   = (const float*)d_in[5];
    const float* dt_bias = (const float*)d_in[6];
    const float* conv_w  = (const float*)d_in[7];
    const float* conv_b  = (const float*)d_in[8];
    const float* norm_w  = (const float*)d_in[9];
    const float* W_out   = (const float*)d_in[10];
    float* out = (float*)d_out;

    float* ws = (float*)d_ws;
    size_t off = 0;
    float* mixed = ws + off; off += (size_t)NTOK * CONV_DIM;       // 33.5M
    float* z     = ws + off; off += (size_t)NTOK * VAL_DIM;        // 16.8M
    float* qn    = ws + off; off += (size_t)NTOK * HKq * DKq;      // 8.4M
    float* kn    = ws + off; off += (size_t)NTOK * HKq * DKq;      // 8.4M
    float* vv    = ws + off; off += (size_t)NTOK * HVq * DVq;      // 16.8M
    float* gatep = ws + off; off += (size_t)NTOK * HVq;            // 131K
    float* betap = ws + off; off += (size_t)NTOK * HVq;            // 131K
    float* attn  = ws + off; off += (size_t)NTOK * HVq * DVq;      // 16.8M
    float* hg    = ws + off; off += (size_t)NTOK * VAL_DIM;        // 16.8M

    // 1) mixed_qkv = x @ W_qkv      [4096 x 8192]
    gemm_bf16_wmma<<<dim3(CONV_DIM / BN, NTOK / BM), 256, 0, stream>>>(
        x, W_qkv, mixed, NTOK, CONV_DIM, Dq);
    // 2) z = x @ W_z                [4096 x 4096]
    gemm_bf16_wmma<<<dim3(VAL_DIM / BN, NTOK / BM), 256, 0, stream>>>(
        x, W_z, z, NTOK, VAL_DIM, Dq);
    // 3) beta / gate
    proj_gate<<<NTOK / 8, 256, 0, stream>>>(x, W_b, W_a, A_log, dt_bias, betap, gatep);
    // 4) conv + SiLU + split + L2 norm
    conv_silu_split<<<dim3(CONV_DIM / 128, Lq, Bq), 128, 0, stream>>>(
        mixed, conv_w, conv_b, qn, kn, vv);
    // 5) gated delta rule scan (64 sequences)
    gdr_scan<<<Bq * HVq, 256, 0, stream>>>(qn, kn, vv, gatep, betap, attn);
    // 6) RMS norm + SiLU(z) gate
    norm_gate<<<NTOK * HVq, 128, 0, stream>>>(attn, z, norm_w, hg);
    // 7) out = hg @ W_out           [4096 x 2048], K = 4096
    gemm_bf16_wmma<<<dim3(Dq / BN, NTOK / BM), 256, 0, stream>>>(
        hg, W_out, out, NTOK, Dq, VAL_DIM);
}
